// action_net_gnn_stream_67774583931073
// MI455X (gfx1250) — matrix operand order repaired and checked by reading, not compiled, not run on hardware
//
#include <hip/hip_runtime.h>
#include <math.h>

// ---------------------------------------------------------------------------
// MI455X (gfx1250) GNN pipeline: concat -> 3x(GraphConv+ReLU) -> score ->
// top-k (bitonic sort, 256KB LDS) -> scale/relu -> classifier.
// GEMMs use v_wmma_f32_16x16x32_bf16 (bf16 in, f32 accumulate) with
// double-buffered async global->LDS weight staging (ASYNCcnt path).
// ---------------------------------------------------------------------------

typedef __attribute__((ext_vector_type(16))) __bf16    v16bf;
typedef __attribute__((ext_vector_type(8)))  float     v8f;
typedef __attribute__((ext_vector_type(8)))  unsigned  v8u;
typedef __attribute__((ext_vector_type(4)))  int       v4i;

#define NNODES 32768
#define DMODEL 256
#define KTOP   16384
#define NCLS   11

#if defined(__has_builtin)
#if __has_builtin(__builtin_amdgcn_global_load_async_to_lds_b128)
#define USE_ASYNC_LDS 1
#endif
#endif

typedef __attribute__((address_space(1))) v4i* g4p;   // global int4*
typedef __attribute__((address_space(3))) v4i* l4p;   // LDS int4*

__device__ __forceinline__ void wait_async0() {
#if defined(__has_builtin)
#if __has_builtin(__builtin_amdgcn_s_wait_asynccnt)
    __builtin_amdgcn_s_wait_asynccnt(0);
    return;
#else
    asm volatile("s_wait_asynccnt 0" ::: "memory");
    return;
#endif
#else
    asm volatile("s_wait_asynccnt 0" ::: "memory");
#endif
}

__device__ __forceinline__ unsigned short f2bfu(float f) {   // RNE (pack kernel)
    unsigned u = __builtin_bit_cast(unsigned, f);
    u += 0x7FFFu + ((u >> 16) & 1u);
    return (unsigned short)(u >> 16);
}

// Truncating pack of two fp32 -> two bf16 in one v_perm_b32:
// result = { hi[31:16], lo[31:16] }
__device__ __forceinline__ unsigned pkbf(float lo, float hi) {
    return __builtin_amdgcn_perm(__builtin_bit_cast(unsigned, hi),
                                 __builtin_bit_cast(unsigned, lo), 0x07060302u);
}

// ---------------------------------------------------------------------------
// x[n, 0:128] = f0 ; x[n, 128:256] = f1
__global__ void k_concat(const float* __restrict__ f0, const float* __restrict__ f1,
                         float* __restrict__ x) {
    int t = blockIdx.x * 256 + threadIdx.x;
    int n = t >> 8, j = t & 255;
    x[t] = (j < 128) ? f0[n * 128 + j] : f1[n * 128 + (j - 128)];
}

// ---------------------------------------------------------------------------
__global__ void k_zero(float4* __restrict__ p, int n4) {
    int t = blockIdx.x * 256 + threadIdx.x;
    if (t < n4) p[t] = make_float4(0.f, 0.f, 0.f, 0.f);
}

// ---------------------------------------------------------------------------
// agg[dst] += x[src] : 64 lanes per edge, float4 gather + 4 fp32 atomics.
// x/agg working sets are L2-resident (32MB each vs 192MB L2).
__global__ void k_edge(const int* __restrict__ src, const int* __restrict__ dst,
                       const float* __restrict__ x, float* __restrict__ agg, int E) {
    int t = blockIdx.x * 256 + threadIdx.x;
    int e = t >> 6;
    if (e >= E) return;
    int q = (t & 63) * 4;
    int s = src[e], d = dst[e];
    float4 v = *(const float4*)(x + (size_t)s * DMODEL + q);
    float* ap = agg + (size_t)d * DMODEL + q;
    __hip_atomic_fetch_add(ap + 0, v.x, __ATOMIC_RELAXED, __HIP_MEMORY_SCOPE_AGENT);
    __hip_atomic_fetch_add(ap + 1, v.y, __ATOMIC_RELAXED, __HIP_MEMORY_SCOPE_AGENT);
    __hip_atomic_fetch_add(ap + 2, v.z, __ATOMIC_RELAXED, __HIP_MEMORY_SCOPE_AGENT);
    __hip_atomic_fetch_add(ap + 3, v.w, __ATOMIC_RELAXED, __HIP_MEMORY_SCOPE_AGENT);
}

// ---------------------------------------------------------------------------
// Pack Wr/Wn (fp32 [256,256] row-major [k][n]) into bf16 WMMA B-fragment
// order: out[mat][kk][nt][lane][j]; element j of lane l maps to
// k = kk*32 + (j>>3)*16 + (l>>4)*8 + (j&7) ; n = nt*16 + (l&15)   (32x16 bf16 B tile).
__global__ void k_packw(const float* __restrict__ Wr, const float* __restrict__ Wn,
                        unsigned short* __restrict__ out) {
    int t = blockIdx.x * 256 + threadIdx.x;   // 0 .. 131071
    int mat  = t >> 16;
    int e    = t & 65535;
    int j    = e & 15;
    int lane = (e >> 4) & 31;
    int nt   = (e >> 9) & 15;
    int kk   = e >> 13;
    int k = kk * 32 + (j >> 3) * 16 + (lane >> 4) * 8 + (j & 7);
    int n = nt * 16 + (lane & 15);
    const float* W = mat ? Wn : Wr;
    out[t] = f2bfu(W[k * 256 + n]);
}

// ---------------------------------------------------------------------------
// 16x32 bf16 WMMA A-fragment from an fp32 row segment (truncating casts,
// one v_perm_b32 per bf16 pair).
// ISA layout: lanes 0-15 (half=0): j0..7 -> K0..7, j8..15 -> K16..23
//             lanes 16-31 (half=1): j0..7 -> K8..15, j8..15 -> K24..31
__device__ __forceinline__ v16bf load_a_frag(const float* __restrict__ row, int half) {
    const float4* p0 = (const float4*)(row + half * 8);
    const float4* p1 = (const float4*)(row + 16 + half * 8);
    float4 a0 = p0[0], a1 = p0[1], a2 = p1[0], a3 = p1[1];
    v8u u;
    u[0] = pkbf(a0.x, a0.y); u[1] = pkbf(a0.z, a0.w);
    u[2] = pkbf(a1.x, a1.y); u[3] = pkbf(a1.z, a1.w);
    u[4] = pkbf(a2.x, a2.y); u[5] = pkbf(a2.z, a2.w);
    u[6] = pkbf(a3.x, a3.y); u[7] = pkbf(a3.z, a3.w);
    return __builtin_bit_cast(v16bf, u);
}

// ---------------------------------------------------------------------------
// Stage one K-step's Wr+Wn fragment tiles (16KB each) into LDS.
#ifdef USE_ASYNC_LDS
__device__ __forceinline__ void stage_async(const unsigned short* __restrict__ Wpk,
                                            int kk, __bf16* dst, int tid) {
    unsigned short* sR = (unsigned short*)(Wpk + kk * 8192);
    unsigned short* sN = (unsigned short*)(Wpk + 65536 + kk * 8192);
    char* dR = (char*)dst;
    char* dN = (char*)(dst + 8192);
    #pragma unroll
    for (int q = 0; q < 4; q++) {
        int o = (tid + q * 256) * 16;   // byte offset, 16B per lane per issue
        __builtin_amdgcn_global_load_async_to_lds_b128(
            (g4p)((char*)sR + o), (l4p)(dR + o), 0, 0);
        __builtin_amdgcn_global_load_async_to_lds_b128(
            (g4p)((char*)sN + o), (l4p)(dN + o), 0, 0);
    }
}
#else
__device__ __forceinline__ void stage_sync(const unsigned short* __restrict__ Wpk,
                                           int kk, __bf16* dst, int tid) {
    const uint4* sR = (const uint4*)(Wpk + kk * 8192);
    const uint4* sN = (const uint4*)(Wpk + 65536 + kk * 8192);
    uint4* dR = (uint4*)dst;
    uint4* dN = (uint4*)(dst + 8192);
    #pragma unroll
    for (int q = 0; q < 4; q++) {
        dR[tid + q * 256] = sR[tid + q * 256];
        dN[tid + q * 256] = sN[tid + q * 256];
    }
}
#endif

// ---------------------------------------------------------------------------
// Y = relu(X@Wr + G@Wn + bias) ; X,G fp32 [32768,256], W packed bf16 frags.
// Block = 256 thr = 8 waves; wave w owns rows [blk*128 + w*16, +16), all 256
// cols: 16 x (16x16 f32) accumulators. Double-buffered LDS weight staging.
__global__ void __launch_bounds__(256) k_gemm(const float* __restrict__ X,
                                              const float* __restrict__ G,
                                              const unsigned short* __restrict__ Wpk,
                                              const float* __restrict__ bias,
                                              float* __restrict__ Y) {
#ifdef USE_ASYNC_LDS
    __shared__ __align__(32) __bf16 Bsm[2][2 * 8192];   // 64KB, double buffered
#else
    __shared__ __align__(32) __bf16 Bsm[1][2 * 8192];   // 32KB, single buffer
#endif
    const int tid  = threadIdx.x;
    const int l    = tid & 31;
    const int wave = tid >> 5;
    const int m    = l & 15;
    const int half = l >> 4;
    const int rowbase = blockIdx.x * 128 + wave * 16;

    v8f acc[16] = {};

#ifdef USE_ASYNC_LDS
    stage_async(Wpk, 0, Bsm[0], tid);
    wait_async0();
    __syncthreads();
#endif

    for (int kk = 0; kk < 8; kk++) {
#ifdef USE_ASYNC_LDS
        __bf16* cur = Bsm[kk & 1];
        if (kk < 7) stage_async(Wpk, kk + 1, Bsm[(kk + 1) & 1], tid);
#else
        __bf16* cur = Bsm[0];
        stage_sync(Wpk, kk, cur, tid);
        __syncthreads();
#endif
        const float* xr = X + (size_t)(rowbase + m) * DMODEL + kk * 32;
        const float* gr = G + (size_t)(rowbase + m) * DMODEL + kk * 32;
        v16bf aX = load_a_frag(xr, half);
        v16bf aG = load_a_frag(gr, half);

        #pragma unroll
        for (int nt = 0; nt < 16; nt++) {
            v16bf bR = *(const v16bf*)(cur + (nt * 32 + l) * 16);
            v16bf bN = *(const v16bf*)(cur + 8192 + (nt * 32 + l) * 16);
            acc[nt] = __builtin_amdgcn_wmma_f32_16x16x32_bf16(
                false, aX, false, bR, (short)0, acc[nt], false, false);
            acc[nt] = __builtin_amdgcn_wmma_f32_16x16x32_bf16(
                false, aG, false, bN, (short)0, acc[nt], false, false);
        }
#ifdef USE_ASYNC_LDS
        wait_async0();       // next buffer fully landed in LDS
        __syncthreads();     // everyone done reading cur + done staging next
#else
        __syncthreads();
#endif
    }

    // epilogue: bias + ReLU. C layout: VGPR i -> row (half?8:0)+i, col = l&15.
    #pragma unroll
    for (int nt = 0; nt < 16; nt++) {
        int n = nt * 16 + m;
        float bv = bias[n];
        #pragma unroll
        for (int i = 0; i < 8; i++) {
            float v = acc[nt][i] + bv;
            v = v > 0.f ? v : 0.f;
            Y[(size_t)(rowbase + half * 8 + i) * DMODEL + n] = v;
        }
    }
}

// ---------------------------------------------------------------------------
// score[n] = x[n].Wpr + agg[n].Wpn + bp : one wave32 per node, shfl reduce.
__global__ void k_score(const float* __restrict__ X, const float* __restrict__ G,
                        const float* __restrict__ Wpr, const float* __restrict__ Wpn,
                        const float* __restrict__ bp, float* __restrict__ score) {
    int wave = threadIdx.x >> 5;
    int lane = threadIdx.x & 31;
    int n = blockIdx.x * 8 + wave;
    const float* xr = X + (size_t)n * DMODEL;
    const float* gr = G + (size_t)n * DMODEL;
    float s = 0.f;
    #pragma unroll
    for (int j = 0; j < 8; j++) {
        int k = j * 32 + lane;
        s += xr[k] * Wpr[k] + gr[k] * Wpn[k];
    }
    #pragma unroll
    for (int mk = 16; mk > 0; mk >>= 1) s += __shfl_xor(s, mk, 32);
    if (lane == 0) score[n] = s + bp[0];
}

// ---------------------------------------------------------------------------
// Full descending bitonic sort of 32768 (score, idx) pairs in 256KB LDS
// (CDNA5: 320KB/WGP). Single 1024-thread workgroup; tie-break on index.
__global__ void k_sort(const float* __restrict__ score,
                       float* __restrict__ skey, int* __restrict__ sidx) {
    extern __shared__ char smem[];
    float* keys = (float*)smem;
    int*   idx  = (int*)(smem + NNODES * sizeof(float));
    const int tid = threadIdx.x;   // 1024 threads
    for (int r = 0; r < NNODES / 1024; r++) {
        int i = r * 1024 + tid;
        keys[i] = score[i];
        idx[i]  = i;
    }
    __syncthreads();
    for (int size = 2; size <= NNODES; size <<= 1) {
        for (int stride = size >> 1; stride > 0; stride >>= 1) {
            for (int r = 0; r < NNODES / 2 / 1024; r++) {
                int i   = r * 1024 + tid;
                int pos = 2 * i - (i & (stride - 1));
                int par = pos + stride;
                bool dec = ((pos & size) == 0);   // overall descending
                float k1 = keys[pos], k2 = keys[par];
                int   i1 = idx[pos],  i2 = idx[par];
                bool sw = dec ? ((k1 < k2) || (k1 == k2 && i1 > i2))
                              : ((k1 > k2) || (k1 == k2 && i1 < i2));
                if (sw) { keys[pos] = k2; keys[par] = k1; idx[pos] = i2; idx[par] = i1; }
            }
            __syncthreads();
        }
    }
    for (int r = 0; r < NNODES / 1024; r++) {
        int i = r * 1024 + tid;
        skey[i] = keys[i];
        sidx[i] = idx[i];
    }
}

// ---------------------------------------------------------------------------
// logits[i,c] = sum_k relu(x[perm[i],k]*tanh(s_i)) * Wc[k,c] + bc[c]
__global__ void k_cls(const float* __restrict__ X, const float* __restrict__ skey,
                      const int* __restrict__ sidx, const float* __restrict__ Wc,
                      const float* __restrict__ bc, float* __restrict__ out, int total) {
    int t = blockIdx.x * 256 + threadIdx.x;
    if (t >= total) return;
    int i = t / NCLS, c = t % NCLS;
    int n = sidx[i];
    float sc = tanhf(skey[i]);
    const float* xr = X + (size_t)n * DMODEL;
    float acc = bc[c];
    for (int k = 0; k < DMODEL; k++) {
        float v = xr[k] * sc;
        v = v > 0.f ? v : 0.f;
        acc += v * Wc[k * NCLS + c];
    }
    out[t] = acc;
}

// ---------------------------------------------------------------------------
extern "C" void kernel_launch(void* const* d_in, const int* in_sizes, int n_in,
                              void* d_out, int out_size, void* d_ws, size_t ws_size,
                              hipStream_t stream) {
    const float* f0  = (const float*)d_in[0];
    const float* f1  = (const float*)d_in[1];
    const int*   ei  = (const int*)d_in[2];
    const float* Wr  = (const float*)d_in[3];
    const float* Wn  = (const float*)d_in[4];
    const float* b   = (const float*)d_in[5];
    const float* Wpr = (const float*)d_in[6];
    const float* Wpn = (const float*)d_in[7];
    const float* bp  = (const float*)d_in[8];
    const float* Wc  = (const float*)d_in[9];
    const float* bc  = (const float*)d_in[10];
    float* out = (float*)d_out;
    const int E = in_sizes[2] / 2;
    (void)n_in; (void)out_size; (void)ws_size;

    char* ws = (char*)d_ws;
    size_t off = 0;
    auto take = [&](size_t bytes) -> void* {
        void* p = ws + off;
        off += (bytes + 255) & ~(size_t)255;
        return p;
    };
    float* xa            = (float*)take((size_t)NNODES * DMODEL * 4);
    float* xb            = (float*)take((size_t)NNODES * DMODEL * 4);
    float* agg           = (float*)take((size_t)NNODES * DMODEL * 4);
    unsigned short* wpk  = (unsigned short*)take(2 * 65536 * 2);
    float* score         = (float*)take((size_t)NNODES * 4);
    float* skey          = (float*)take((size_t)NNODES * 4);
    int*   sidx          = (int*)take((size_t)NNODES * 4);

    const int zeroBlocks = NNODES * DMODEL / 4 / 256;   // float4 stores
    const int edgeBlocks = (E * 64 + 255) / 256;

    k_concat<<<NNODES, 256, 0, stream>>>(f0, f1, xa);

    float* xc = xa; float* xn = xb;
    for (int i = 0; i < 3; i++) {
        k_packw<<<512, 256, 0, stream>>>(Wr + (size_t)i * 65536, Wn + (size_t)i * 65536, wpk);
        k_zero<<<zeroBlocks, 256, 0, stream>>>((float4*)agg, NNODES * DMODEL / 4);
        k_edge<<<edgeBlocks, 256, 0, stream>>>(ei, ei + E, xc, agg, E);
        k_gemm<<<NNODES / 128, 256, 0, stream>>>(xc, agg, wpk, b + (size_t)i * DMODEL, xn);
        float* tmp = xc; xc = xn; xn = tmp;
    }

    // pooling attention score (one more aggregation pass over final x)
    k_zero<<<zeroBlocks, 256, 0, stream>>>((float4*)agg, NNODES * DMODEL / 4);
    k_edge<<<edgeBlocks, 256, 0, stream>>>(ei, ei + E, xc, agg, E);
    k_score<<<NNODES / 8, 256, 0, stream>>>(xc, agg, Wpr, Wpn, bp, score);

    // exact top-k via full descending bitonic sort (k = N/2)
    k_sort<<<1, 1024, NNODES * 8, stream>>>(score, skey, sidx);

    const int total = KTOP * NCLS;
    k_cls<<<(total + 255) / 256, 256, 0, stream>>>(xc, skey, sidx, Wc, bc, out, total);
}
